// RhythmMamba_sda_16157666968350
// MI455X (gfx1250) — compile-verified, matrix operand order checked by reading, not run on hardware
//
#include <hip/hip_runtime.h>
#include <hip/hip_bf16.h>
#include <math.h>
#include <stdint.h>

// ---------------- problem constants (from reference) ----------------
#define BATCH 2
#define LSEQ  512
#define DIM   192
#define DST   48          // D_STATE
#define DCONV 4
#define DI    384         // DIM*EXPAND
#define DTR   12
#define HID   384
#define SEG   4
#define TT    128         // LSEQ/SEG
#define NB    (SEG*BATCH) // 8 expanded batches
#define M1    (NB*LSEQ)   // 4096 rows for mamba GEMMs
#define M2    (BATCH*LSEQ)// 1024 rows for ffn GEMMs
#define DBLW  (DTR + 2*DST) // 108
#define EPSF  1e-5f

typedef __attribute__((ext_vector_type(2)))  float    v2f;
typedef __attribute__((ext_vector_type(8)))  float    v8f;
typedef __attribute__((ext_vector_type(16))) _Float16 v16h;

#if defined(__gfx1250__) && __has_builtin(__builtin_amdgcn_wmma_f32_16x16x4_f32)
#define USE_F32_WMMA 1
#else
#define USE_F32_WMMA 0
#endif

#define AS1 __attribute__((address_space(1)))
#define AS3 __attribute__((address_space(3)))

// exact pointee types expected by the async builtins (from compiler diagnostics):
//   b128: 'int __attribute__((vector_size(16)))' ; b32: 'int'
typedef int a_v4i __attribute__((__vector_size__(4 * sizeof(int))));

// CDNA5 async global->LDS copies (ASYNCcnt-tracked). Guarded so absence of the
// builtins degrades to sync b128 loads instead of breaking the build.
#if defined(__AMDGCN__) &&                                            \
    __has_builtin(__builtin_amdgcn_global_load_async_to_lds_b128) && \
    __has_builtin(__builtin_amdgcn_global_load_async_to_lds_b32)  && \
    __has_builtin(__builtin_amdgcn_s_wait_asynccnt)
#define HAVE_ASYNC 1
#else
#define HAVE_ASYNC 0
#endif

__device__ __forceinline__ void g2l_b128(const float* g, float* l) {
#if HAVE_ASYNC
  // flat LDS address truncates to the 32-bit LDS offset (aperture rule), so the
  // low 32 bits of the generic pointer are the AS3 address.
  __builtin_amdgcn_global_load_async_to_lds_b128(
      (AS1 a_v4i*)(uintptr_t)g, (AS3 a_v4i*)(uint32_t)(uintptr_t)l, 0, 0);
#else
  *(float4*)l = *(const float4*)g;
#endif
}
__device__ __forceinline__ void g2l_b32(const float* g, float* l) {
#if HAVE_ASYNC
  __builtin_amdgcn_global_load_async_to_lds_b32(
      (AS1 int*)(uintptr_t)g, (AS3 int*)(uint32_t)(uintptr_t)l, 0, 0);
#else
  *l = *g;
#endif
}
__device__ __forceinline__ void g2l_fence() {
#if HAVE_ASYNC
  __builtin_amdgcn_s_wait_asynccnt(0);
#endif
}

// ---------------- generic WMMA GEMM ----------------
// C[m,n] = epi( alpha * sum_k A[m,k]*B[n,k] (+ beta*Csrc) (*scale[n]) (+bias[n]) ) (+postadd)
// A: row-major, lda.  B: element (n,k) at n*ldbn + k*ldbk (weights: ldbn=K,ldbk=1;
// DFT right-operand: ldbn=1, ldbk=cols).  Block = 8 waves -> 32x64 output tile.
#define KT 32
// LDS row stride 36 floats: 16B-aligned quads for b128 stores, and 36*r mod 64
// distinct for r=0..15 -> conflict-free column reads during WMMA operand fetch.
#define LDP 36
enum { EPI_NONE = 0, EPI_RELU = 1, EPI_SOFTPLUS = 2 };

template <int EPI>
__global__ __launch_bounds__(256) void wmma_gemm_f32(
    const float* __restrict__ A, int lda, long long batchA,
    const float* __restrict__ B, int ldbn, int ldbk, long long batchB,
    float* __restrict__ C, int ldc, long long batchC,
    const float* __restrict__ Csrc, float beta,
    const float* __restrict__ scale, const float* __restrict__ bias,
    const float* __restrict__ postadd,
    float alpha, int M, int N, int K) {
  __shared__ float As[32][LDP];
  __shared__ float Bs[64][LDP];
  const int tid  = threadIdx.x;
  const int wave = tid >> 5, lane = tid & 31;
  const int wm = wave >> 2, wn = wave & 3;     // 2x4 wave grid
  const int half = lane >> 4, l16 = lane & 15;
  const int m0 = blockIdx.y * 32, n0 = blockIdx.x * 64;
  const long long z = blockIdx.z;
  A += batchA * z;  B += batchB * z;  C += batchC * z;
  const float* Cs2 = Csrc ? Csrc + batchC * z : nullptr;
  const float* Pad = postadd ? postadd + batchC * z : nullptr;

  const bool fullM = (m0 + 32 <= M);
  const bool fullN = (n0 + 64 <= N);
  const bool bVecK = (ldbk == 1) && ((ldbn & 3) == 0);
  const bool bVecN = (ldbn == 1);

  v8f acc = {};
  for (int k0 = 0; k0 < K; k0 += KT) {
    const bool fullK = (k0 + KT <= K);
    // ---- stage A tile (32 x KT) ----
    if (fullM && fullK) {
      const int r = tid >> 3, q = (tid & 7) * 4;   // one float4 per thread
      const float* gp = A + (long long)(m0 + r) * lda + (k0 + q);
      g2l_b128(gp, &As[r][q]);
      if (k0 + KT < K) __builtin_prefetch(gp + KT, 0, 3);
    } else {
      for (int e = tid; e < 32 * KT; e += 256) {
        const int r = e >> 5, k = e & (KT - 1);
        const int gm = m0 + r, gk = k0 + k;
        As[r][k] = (gm < M && gk < K) ? A[(long long)gm * lda + gk] : 0.f;
      }
    }
    // ---- stage B tile (64 x KT) ----
    if (fullN && fullK && bVecK) {
      // weights / contiguous-k operand: vector loads along k
      for (int e = tid; e < 64 * (KT / 4); e += 256) {
        const int r = e >> 3, q = (e & 7) * 4;
        g2l_b128(B + (long long)(n0 + r) * ldbn + (k0 + q), &Bs[r][q]);
      }
    } else if (fullN && fullK && bVecN) {
      // DFT right-operand: contiguous in n, strided in k -> per-element gather
      for (int e = tid; e < 64 * KT; e += 256) {
        const int n = e & 63, k = e >> 6;   // consecutive tid -> consecutive n
        g2l_b32(B + (n0 + n) + (long long)(k0 + k) * ldbk, &Bs[n][k]);
      }
    } else {
      for (int e = tid; e < 64 * KT; e += 256) {
        const int r = e >> 5, k = e & (KT - 1);
        const int gn = n0 + r, gk = k0 + k;
        Bs[r][k] = (gn < N && gk < K)
                       ? B[(long long)gn * ldbn + (long long)gk * ldbk] : 0.f;
      }
    }
    g2l_fence();
    __syncthreads();
#if USE_F32_WMMA
#pragma unroll
    for (int kk = 0; kk < KT; kk += 4) {
      v2f a, b;
      const int ks = kk + half * 2;
      a.x = As[wm * 16 + l16][ks];  a.y = As[wm * 16 + l16][ks + 1];
      b.x = Bs[wn * 16 + l16][ks];  b.y = Bs[wn * 16 + l16][ks + 1];
      acc = __builtin_amdgcn_wmma_f32_16x16x4_f32(false, a, false, b,
                                                  (short)0, acc, false, false);
    }
#else
    // f16 fallback: one 16x16x32 WMMA per K-tile (A 16x32 / B 32x16 layouts)
    v16h a16, b16;
#pragma unroll
    for (int v = 0; v < 8; ++v) {
      const int kb = (v < 4) ? (half * 8 + 2 * v) : (16 + half * 8 + 2 * (v - 4));
      a16[2 * v]     = (_Float16)As[wm * 16 + l16][kb];
      a16[2 * v + 1] = (_Float16)As[wm * 16 + l16][kb + 1];
      b16[2 * v]     = (_Float16)Bs[wn * 16 + l16][kb];
      b16[2 * v + 1] = (_Float16)Bs[wn * 16 + l16][kb + 1];
    }
    acc = __builtin_amdgcn_wmma_f32_16x16x32_f16(false, a16, false, b16,
                                                 (short)0, acc, false, false);
#endif
    __syncthreads();
  }
#pragma unroll
  for (int r = 0; r < 8; ++r) {
    const int row = m0 + wm * 16 + r + (half ? 8 : 0);
    const int col = n0 + wn * 16 + l16;
    if (row < M && col < N) {
      float v = alpha * acc[r];
      if (Cs2)   v += beta * Cs2[(long long)row * ldc + col];
      if (scale) v *= scale[col];
      if (bias)  v += bias[col];
      if (EPI == EPI_RELU)          v = fmaxf(v, 0.f);
      else if (EPI == EPI_SOFTPLUS) v = (v > 20.f) ? v : log1pf(__expf(v));
      if (Pad)   v += Pad[(long long)row * ldc + col];
      C[(long long)row * ldc + col] = v;
    }
  }
}

// ---------------- elementwise / scan kernels ----------------

// LN over DIM with the 4-way time-shift gather (mamba input). One wave per row.
__global__ __launch_bounds__(256) void ln0_shift_kernel(
    const float* __restrict__ x, const float* __restrict__ g,
    const float* __restrict__ b, float* __restrict__ out) {
  const int row = blockIdx.x * 8 + (threadIdx.x >> 5);
  if (row >= M1) return;
  const int lane = threadIdx.x & 31;
  const int s = row >> 9, t = row & (LSEQ - 1);
  const int i = s >> 1, bb = s & 1;
  const int tsrc = (t < LSEQ - i * TT) ? (t + i * TT) : t;
  const float* p = x + ((long long)bb * LSEQ + tsrc) * DIM;
  float sum = 0.f, sq = 0.f;
  for (int c = lane; c < DIM; c += 32) { float v = p[c]; sum += v; sq += v * v; }
  for (int off = 16; off; off >>= 1) {
    sum += __shfl_xor(sum, off, 32);
    sq  += __shfl_xor(sq,  off, 32);
  }
  const float mean = sum / DIM;
  const float inv  = rsqrtf(sq / DIM - mean * mean + EPSF);
  float* o = out + (long long)row * DIM;
  for (int c = lane; c < DIM; c += 32)
    o[c] = (p[c] - mean) * inv * g[c] + b[c];
}

// Generic LN over DIM, optional residual add. One wave per row.
__global__ __launch_bounds__(256) void ln_kernel(
    const float* __restrict__ src, const float* __restrict__ g,
    const float* __restrict__ b, const float* __restrict__ resid,
    float* __restrict__ out, int rows) {
  const int row = blockIdx.x * 8 + (threadIdx.x >> 5);
  if (row >= rows) return;
  const int lane = threadIdx.x & 31;
  const float* p = src + (long long)row * DIM;
  float sum = 0.f, sq = 0.f;
  for (int c = lane; c < DIM; c += 32) { float v = p[c]; sum += v; sq += v * v; }
  for (int off = 16; off; off >>= 1) {
    sum += __shfl_xor(sum, off, 32);
    sq  += __shfl_xor(sq,  off, 32);
  }
  const float mean = sum / DIM;
  const float inv  = rsqrtf(sq / DIM - mean * mean + EPSF);
  for (int c = lane; c < DIM; c += 32) {
    float v = (p[c] - mean) * inv * g[c] + b[c];
    if (resid) v += resid[(long long)row * DIM + c];
    out[(long long)row * DIM + c] = v;
  }
}

// Depthwise causal conv (k=4) over time + SiLU; input is xz[...,:DI] (stride 2*DI).
__global__ __launch_bounds__(256) void conv_silu_kernel(
    const float* __restrict__ xz, const float* __restrict__ cw,
    const float* __restrict__ cb, float* __restrict__ xm) {
  const int idx = blockIdx.x * 256 + threadIdx.x;
  if (idx >= NB * LSEQ * DI) return;
  const int c = idx % DI, t = (idx / DI) % LSEQ, s = idx / (DI * LSEQ);
  const float* base = xz + (long long)s * LSEQ * (2 * DI) + c;
  float acc = cb[c];
#pragma unroll
  for (int j = 0; j < DCONV; ++j) {
    const int ts = t - (DCONV - 1) + j;
    if (ts >= 0) acc += base[(long long)ts * (2 * DI)] * cw[c * DCONV + j];
  }
  xm[idx] = acc / (1.f + __expf(-acc));   // silu
}

// Selective scan: one thread per channel, 48 states in registers, B/C in LDS.
__global__ __launch_bounds__(DI) void scan_kernel(
    const float* __restrict__ delta, const float* __restrict__ xm,
    const float* __restrict__ dbl, const float* __restrict__ A_log,
    float* __restrict__ ys) {
  const int s = blockIdx.x;     // expanded batch 0..7
  const int c = threadIdx.x;    // channel 0..383
  __shared__ float Bs[DST], Cs[DST];
  float h[DST], Ac[DST];
#pragma unroll
  for (int j = 0; j < DST; ++j) { Ac[j] = -__expf(A_log[c * DST + j]); h[j] = 0.f; }
  const float* db = dbl + (long long)s * LSEQ * DBLW;
  const long long rowbase = (long long)s * LSEQ;
  for (int t = 0; t < LSEQ; ++t) {
    if (c < DST)            Bs[c]       = db[t * DBLW + DTR + c];
    else if (c < 2 * DST)   Cs[c - DST] = db[t * DBLW + DTR + DST + (c - DST)];
    __syncthreads();
    const long long r = (rowbase + t) * DI + c;
    const float d = delta[r], u = xm[r];
    const float du = d * u;
    float y = 0.f;
#pragma unroll
    for (int j = 0; j < DST; ++j) {
      h[j] = __expf(d * Ac[j]) * h[j] + du * Bs[j];
      y += h[j] * Cs[j];
    }
    ys[r] = y;
    __syncthreads();
  }
}

// y = (ys + xm*D) * silu(z) ; z = xz[...,DI:]
__global__ __launch_bounds__(256) void gate_kernel(
    const float* __restrict__ ys, const float* __restrict__ xm,
    const float* __restrict__ xz, const float* __restrict__ Dp,
    float* __restrict__ g) {
  const int idx = blockIdx.x * 256 + threadIdx.x;
  if (idx >= NB * LSEQ * DI) return;
  const int c = idx % DI;
  const long long row = idx / DI;
  const float z = xz[row * (2 * DI) + DI + c];
  const float sz = z / (1.f + __expf(-z));
  g[idx] = (ys[idx] + xm[idx] * Dp[c]) * sz;
}

// Segment-average recombination: y (8,512,192) -> y0 (2,512,192)
__global__ __launch_bounds__(256) void recombine_kernel(
    const float* __restrict__ y, float* __restrict__ y0) {
  const int idx = blockIdx.x * 256 + threadIdx.x;
  if (idx >= BATCH * LSEQ * DIM) return;
  const int c = idx % DIM, t = (idx / DIM) % LSEQ, bb = idx / (DIM * LSEQ);
  const int i = t >> 7, local = t & (TT - 1);
  float acc = y[(((long long)bb * LSEQ) + t) * DIM + c];
  for (int j = 0; j < i; ++j) {
    const int s  = (j + 1) * BATCH + bb;
    const int ts = TT * (i - j - 1) + local;
    acc += y[(((long long)s * LSEQ) + ts) * DIM + c];
  }
  y0[idx] = acc / (float)(i + 1);
}

// DFT cos/sin matrices, exact angle reduction via (k*n) mod 512.
__global__ __launch_bounds__(256) void dft_init_kernel(
    float* __restrict__ Fc, float* __restrict__ Fs) {
  const int idx = blockIdx.x * 256 + threadIdx.x;
  if (idx >= LSEQ * LSEQ) return;
  const int k = idx >> 9, n = idx & (LSEQ - 1);
  const int m = (k * n) & (LSEQ - 1);
  const float th = (float)m * (6.283185307179586f / (float)LSEQ);
  Fc[idx] = __cosf(th);
  Fs[idx] = __sinf(th);
}

// complex diagonal mix + relu, in place on (Xr,Xi)
__global__ __launch_bounds__(256) void freq_diag_kernel(
    float* __restrict__ Xr, float* __restrict__ Xi,
    const float* __restrict__ r_w, const float* __restrict__ i_w,
    const float* __restrict__ rb, const float* __restrict__ ib) {
  const int idx = blockIdx.x * 256 + threadIdx.x;
  if (idx >= BATCH * LSEQ * HID) return;
  const int c = idx % HID;
  const float dr = r_w[(long long)c * HID + c], di = i_w[(long long)c * HID + c];
  const float a = Xr[idx], b = Xi[idx];
  Xr[idx] = fmaxf(a * dr - b * di + rb[c], 0.f);
  Xi[idx] = fmaxf(b * dr + a * di + ib[c], 0.f);
}

// ---------------- host side ----------------
static void launch_gemm(int epi, const float* A, int lda, long long bA,
                        const float* B, int ldbn, int ldbk, long long bB,
                        float* C, int ldc, long long bC,
                        const float* Csrc, float beta, const float* scale,
                        const float* bias, const float* postadd, float alpha,
                        int M, int N, int K, int batch, hipStream_t s) {
  dim3 grid((N + 63) / 64, (M + 31) / 32, batch), blk(256);
  switch (epi) {
    case EPI_RELU:
      wmma_gemm_f32<EPI_RELU><<<grid, blk, 0, s>>>(A, lda, bA, B, ldbn, ldbk, bB,
          C, ldc, bC, Csrc, beta, scale, bias, postadd, alpha, M, N, K);
      break;
    case EPI_SOFTPLUS:
      wmma_gemm_f32<EPI_SOFTPLUS><<<grid, blk, 0, s>>>(A, lda, bA, B, ldbn, ldbk, bB,
          C, ldc, bC, Csrc, beta, scale, bias, postadd, alpha, M, N, K);
      break;
    default:
      wmma_gemm_f32<EPI_NONE><<<grid, blk, 0, s>>>(A, lda, bA, B, ldbn, ldbk, bB,
          C, ldc, bC, Csrc, beta, scale, bias, postadd, alpha, M, N, K);
      break;
  }
}

extern "C" void kernel_launch(void* const* d_in, const int* in_sizes, int n_in,
                              void* d_out, int out_size, void* d_ws, size_t ws_size,
                              hipStream_t stream) {
  (void)in_sizes; (void)n_in; (void)out_size; (void)ws_size;
  const float* x        = (const float*)d_in[0];
  const float* ln0_w    = (const float*)d_in[1];
  const float* ln0_b    = (const float*)d_in[2];
  const float* in_projw = (const float*)d_in[3];
  const float* conv_w   = (const float*)d_in[4];
  const float* conv_b   = (const float*)d_in[5];
  const float* x_projw  = (const float*)d_in[6];
  const float* dt_w     = (const float*)d_in[7];
  const float* dt_b     = (const float*)d_in[8];
  const float* A_log    = (const float*)d_in[9];
  const float* D_param  = (const float*)d_in[10];
  const float* out_projw= (const float*)d_in[11];
  const float* ln1_w    = (const float*)d_in[12];
  const float* ln1_b    = (const float*)d_in[13];
  const float* ln2_w    = (const float*)d_in[14];
  const float* ln2_b    = (const float*)d_in[15];
  const float* fc1_w    = (const float*)d_in[16];
  const float* bn1_g    = (const float*)d_in[17];
  const float* bn1_b    = (const float*)d_in[18];
  const float* r_w      = (const float*)d_in[19];
  const float* i_w      = (const float*)d_in[20];
  const float* rb       = (const float*)d_in[21];
  const float* ib       = (const float*)d_in[22];
  const float* fc2_w    = (const float*)d_in[23];
  const float* bn2_g    = (const float*)d_in[24];
  const float* bn2_b    = (const float*)d_in[25];
  float* out = (float*)d_out;

  // workspace carve (floats)
  float* W = (float*)d_ws;
  size_t o = 0;
  auto take = [&](size_t n) { float* p = W + o; o += n; return p; };
  float* xo_ln = take((size_t)M1 * DIM);     // 4096x192 LN'd shifted input
  float* xz    = take((size_t)M1 * 2 * DI);  // 4096x768
  float* xm    = take((size_t)M1 * DI);      // 4096x384 conv+silu
  float* dbl   = take((size_t)M1 * DBLW);    // 4096x108
  float* delta = take((size_t)M1 * DI);      // 4096x384 (reused as gate out)
  float* ysb   = take((size_t)M1 * DI);      // scan output
  float* ym    = take((size_t)M1 * DIM);     // mamba out 4096x192
  float* y0    = take((size_t)M2 * DIM);     // 1024x192
  float* x1    = take((size_t)M2 * DIM);
  float* x2    = take((size_t)M2 * DIM);
  float* hbuf  = take((size_t)M2 * HID);     // fc1 out (reused for ifft result)
  float* Xr    = take((size_t)M2 * HID);
  float* Xi    = take((size_t)M2 * HID);
  float* Fc    = take((size_t)LSEQ * LSEQ);
  float* Fs    = take((size_t)LSEQ * LSEQ);

  const float bn_scale = 1.0f / sqrtf(1.0f + EPSF);
  const float invN     = 1.0f / sqrtf((float)LSEQ);   // ortho norm
  const long long LH   = (long long)LSEQ * HID;

  // 0) DFT matrices
  dft_init_kernel<<<(LSEQ * LSEQ + 255) / 256, 256, 0, stream>>>(Fc, Fs);
  // 1) shift + LN0
  ln0_shift_kernel<<<(M1 + 7) / 8, 256, 0, stream>>>(x, ln0_w, ln0_b, xo_ln);
  // 2) in_proj: xz = xo_ln @ in_proj_w.T      (4096x192 x 192x768)
  launch_gemm(EPI_NONE, xo_ln, DIM, 0, in_projw, DIM, 1, 0, xz, 2 * DI, 0,
              nullptr, 0.f, nullptr, nullptr, nullptr, 1.f, M1, 2 * DI, DIM, 1, stream);
  // 3) depthwise conv + silu
  conv_silu_kernel<<<(M1 * DI + 255) / 256, 256, 0, stream>>>(xz, conv_w, conv_b, xm);
  // 4) x_proj: dbl = xm @ x_proj_w.T          (4096x384 x 384x108)
  launch_gemm(EPI_NONE, xm, DI, 0, x_projw, DI, 1, 0, dbl, DBLW, 0,
              nullptr, 0.f, nullptr, nullptr, nullptr, 1.f, M1, DBLW, DI, 1, stream);
  // 5) delta = softplus(dbl[:,:12] @ dt_w.T + dt_b)
  launch_gemm(EPI_SOFTPLUS, dbl, DBLW, 0, dt_w, DTR, 1, 0, delta, DI, 0,
              nullptr, 0.f, nullptr, dt_b, nullptr, 1.f, M1, DI, DTR, 1, stream);
  // 6) selective scan
  scan_kernel<<<NB, DI, 0, stream>>>(delta, xm, dbl, A_log, ysb);
  // 7) gating (overwrites delta buffer)
  gate_kernel<<<(M1 * DI + 255) / 256, 256, 0, stream>>>(ysb, xm, xz, D_param, delta);
  // 8) out_proj: ym = g @ out_proj_w.T        (4096x384 x 384x192)
  launch_gemm(EPI_NONE, delta, DI, 0, out_projw, DI, 1, 0, ym, DIM, 0,
              nullptr, 0.f, nullptr, nullptr, nullptr, 1.f, M1, DIM, DI, 1, stream);
  // 9) segment recombination -> y0
  recombine_kernel<<<(M2 * DIM + 255) / 256, 256, 0, stream>>>(ym, y0);
  // 10) x1 = x + LN1(y0) ; x2 = LN2(x1)
  ln_kernel<<<(M2 + 7) / 8, 256, 0, stream>>>(y0, ln1_w, ln1_b, x, x1, M2);
  ln_kernel<<<(M2 + 7) / 8, 256, 0, stream>>>(x1, ln2_w, ln2_b, nullptr, x2, M2);
  // 11) fc1: h = relu(x2 @ fc1_w.T * bn_scale * bn1_g + bn1_b)
  launch_gemm(EPI_RELU, x2, DIM, 0, fc1_w, DIM, 1, 0, hbuf, HID, 0,
              nullptr, 0.f, bn1_g, bn1_b, nullptr, bn_scale, M2, HID, DIM, 1, stream);
  // 12) forward DFT (ortho): Xr = invN*Fc@h ; Xi = -invN*Fs@h   (batched over 2)
  launch_gemm(EPI_NONE, Fc, LSEQ, 0, hbuf, 1, HID, LH, Xr, HID, LH,
              nullptr, 0.f, nullptr, nullptr, nullptr, invN, LSEQ, HID, LSEQ, BATCH, stream);
  launch_gemm(EPI_NONE, Fs, LSEQ, 0, hbuf, 1, HID, LH, Xi, HID, LH,
              nullptr, 0.f, nullptr, nullptr, nullptr, -invN, LSEQ, HID, LSEQ, BATCH, stream);
  // 13) complex diagonal mix + relu (in place)
  freq_diag_kernel<<<(M2 * HID + 255) / 256, 256, 0, stream>>>(Xr, Xi, r_w, i_w, rb, ib);
  // 14) inverse DFT real part: t = invN*Fc@xr ; t += -invN*Fs@xi
  launch_gemm(EPI_NONE, Fc, LSEQ, 0, Xr, 1, HID, LH, hbuf, HID, LH,
              nullptr, 0.f, nullptr, nullptr, nullptr, invN, LSEQ, HID, LSEQ, BATCH, stream);
  launch_gemm(EPI_NONE, Fs, LSEQ, 0, Xi, 1, HID, LH, hbuf, HID, LH,
              hbuf, 1.f, nullptr, nullptr, nullptr, -invN, LSEQ, HID, LSEQ, BATCH, stream);
  // 15) fc2 + BN fold + residual x1 -> final output
  launch_gemm(EPI_NONE, hbuf, HID, 0, fc2_w, HID, 1, 0, out, DIM, 0,
              nullptr, 0.f, bn2_g, bn2_b, x1, bn_scale, M2, DIM, HID, 1, stream);
}
